// VaeRNNDecoder_37572373905789
// MI455X (gfx1250) — matrix operand order repaired and checked by reading, not compiled, not run on hardware
//
#include <hip/hip_runtime.h>
#include <hip/hip_bf16.h>

// ---------------------------------------------------------------------------
// LSTM decoder (VAE RNN): B=256, H=512, 512 steps.  See round-1 notes.
// Round-2 changes (from gfx1250 asm inspection):
//  * B fragments for all 8 gate tiles load into DISTINCT registers (bf[8])
//    so the 16 global_load_b128 issue as one clause and s_wait_loadcnt
//    staggers per WMMA instead of a full-stop wait before every WMMA.
//  * A fragment for kt+1 prefetched (ds_load) before the WMMA block.
//  * Weight addresses: one per-wave base pointer advanced 1024 B per kt;
//    the 8 tile offsets are compile-time constants (max ~3.1 MB) that fold
//    into the 24-bit signed immediate offset of global_load_b128 —
//    removes the v_lshl_add_u64 per-tile address chains.
// ---------------------------------------------------------------------------

typedef __bf16 bf16_t;
typedef __attribute__((ext_vector_type(16))) __bf16 v16bf;
typedef __attribute__((ext_vector_type(8)))  __bf16 v8bf;
typedef __attribute__((ext_vector_type(8)))  float  v8f;

#define HSZ      512
#define BSZ      256
#define OLEN     512
#define GATE_NT  128   // 2048 / 16 output tiles for gates GEMM
#define GATE_KT  32    // K = 1024 (x|h) / 32
#define LAST_NT  32    // 512 / 16 output tiles for y GEMM
#define LAST_KT  16    // K = 512 / 32
#define LDSTRIDE 1032  // 1024 + 8 bf16 pad (16B aligned, bank-spread)

#define GATE_TILE_ELEMS (GATE_NT * GATE_KT * 512)  // 2,097,152 bf16
#define LAST_TILE_ELEMS (LAST_NT * LAST_KT * 512)  //   262,144 bf16

__device__ __forceinline__ float sigm_f(float x) {
  return 1.0f / (1.0f + __expf(-x));
}
__device__ __forceinline__ float tanh_f(float x) {
  return 2.0f / (1.0f + __expf(-2.0f * x)) - 1.0f;
}

// ---------------------------------------------------------------------------
// Prologue: fp32 weights -> bf16 WMMA-B-fragment tile order (see round 1).
//   n = nt*16 + (lane & 15),  k = kt*32 + e + (lane>=16 ? 16 : 0)
// Gates B is the K-stacked [W_ih^T ; W_hh^T].
// ---------------------------------------------------------------------------
__global__ void lstm_pretile_weights(const float* __restrict__ Wih,
                                     const float* __restrict__ Whh,
                                     const float* __restrict__ Wlast,
                                     bf16_t* __restrict__ ws) {
  int idx = blockIdx.x * blockDim.x + threadIdx.x;
  if (idx < GATE_TILE_ELEMS) {
    int nt   = idx / (GATE_KT * 512);
    int rem  = idx % (GATE_KT * 512);
    int kt   = rem / 512;
    int f    = rem % 512;
    int lane = f >> 4;
    int e    = f & 15;
    int n    = nt * 16 + (lane & 15);
    int k    = kt * 32 + e + ((lane >= 16) ? 16 : 0);
    float v  = (k < HSZ) ? Wih[n * HSZ + k] : Whh[n * HSZ + (k - HSZ)];
    ws[idx]  = (bf16_t)v;
  } else if (idx < GATE_TILE_ELEMS + LAST_TILE_ELEMS) {
    int j    = idx - GATE_TILE_ELEMS;
    int nt   = j / (LAST_KT * 512);
    int rem  = j % (LAST_KT * 512);
    int kt   = rem / 512;
    int f    = rem % 512;
    int lane = f >> 4;
    int e    = f & 15;
    int n    = nt * 16 + (lane & 15);
    int k    = kt * 32 + e + ((lane >= 16) ? 16 : 0);
    ws[idx]  = (bf16_t)Wlast[n * HSZ + k];
  }
}

// ---------------------------------------------------------------------------
// Persistent decoder: blockIdx.x = batch tile (16 rows), 16 waves x 32.
// ---------------------------------------------------------------------------
__global__ __launch_bounds__(512)
void lstm_decode_persistent(const float* __restrict__ z,
                            const float* __restrict__ b_ih,
                            const float* __restrict__ b_hh,
                            const float* __restrict__ b_last,
                            const bf16_t* __restrict__ wt,
                            float* __restrict__ out) {
  __shared__ bf16_t xh[16][LDSTRIDE];  // [row][0..511]=x, [512..1023]=h

  const int tid    = threadIdx.x;
  const int wave   = tid >> 5;
  const int lane   = tid & 31;
  const int lcol   = lane & 15;              // N within a 16-wide tile
  const int rowOff = (lane >= 16) ? 8 : 0;   // C/D: vgpr r -> row r(+8)
  const int aoff   = (lane >= 16) ? 16 : 0;  // A frag K sub-offset
  const int b0     = blockIdx.x * 16;

  // Per-wave weight base pointers (tile offsets become load immediates).
  // Gate tile j=(gg,tt): elem off = (gg*32 + tt)*GATE_KT*512 from gbase.
  const bf16_t* const gbase =
      wt + (size_t)(wave * 2) * GATE_KT * 512 + lane * 16;
  const bf16_t* const lbase =
      wt + GATE_TILE_ELEMS + (size_t)(wave * 2) * LAST_KT * 512 + lane * 16;

  // --- biases (column-only dependent: one scalar per lane per tile) ---
  float gbias[8];
#pragma unroll
  for (int j = 0; j < 8; ++j) {
    int gg = j >> 1, tt = j & 1;
    int col = (gg * 32 + wave * 2 + tt) * 16 + lcol;   // col in [0,2048)
    gbias[j] = b_ih[col] + b_hh[col];
  }
  float ybias[2];
#pragma unroll
  for (int j = 0; j < 2; ++j)
    ybias[j] = b_last[(wave * 2 + j) * 16 + lcol];

  // --- init: x = h = z (bf16 in LDS), c = z (fp32 in registers) ---
  for (int idx = tid; idx < 16 * 512; idx += 512) {
    int mm = idx >> 9, col = idx & 511;
    bf16_t zb = (bf16_t)z[(b0 + mm) * HSZ + col];
    xh[mm][col]       = zb;
    xh[mm][512 + col] = zb;
  }
  float creg[2][8];
#pragma unroll
  for (int tt = 0; tt < 2; ++tt) {
    int col = (wave * 2 + tt) * 16 + lcol;             // hidden col
#pragma unroll
    for (int r = 0; r < 8; ++r)
      creg[tt][r] = z[(b0 + r + rowOff) * HSZ + col];
  }
  __syncthreads();

  const int m = lcol;  // A-fragment row for this lane
  const bf16_t* const arow = &xh[m][aoff];  // A frag base in LDS

#define LOAD_A(K0)                                                         \
  __builtin_shufflevector(*(const v8bf*)(arow + (K0)),                     \
                          *(const v8bf*)(arow + (K0) + 8), 0, 1, 2, 3, 4,  \
                          5, 6, 7, 8, 9, 10, 11, 12, 13, 14, 15)

  for (int t = 0; t < OLEN; ++t) {
    // ---------------- phase 1: gates = [x|h] @ Wg + b ----------------
    v8f acc[8];
#pragma unroll
    for (int j = 0; j < 8; ++j)
#pragma unroll
      for (int r = 0; r < 8; ++r) acc[j][r] = gbias[j];

    const bf16_t* p = gbase;           // advances 512 elems (1024 B) per kt
    v16bf a = LOAD_A(0);
#pragma unroll 2
    for (int kt = 0; kt < GATE_KT; ++kt) {
      // all 8 B fragments -> distinct regs: loads clause, waits stagger
      v16bf bf[8];
#pragma unroll
      for (int j = 0; j < 8; ++j) {
        // compile-time tile offset (<= 3.1 MB, fits 24-bit signed imm)
        bf[j] = *(const v16bf*)(p + ((j >> 1) * 32 + (j & 1)) * (GATE_KT * 512));
      }
      // prefetch next A fragment from LDS before the WMMA block
      v16bf anext = LOAD_A(((kt + 1) & (GATE_KT - 1)) * 32);
#pragma unroll
      for (int j = 0; j < 8; ++j) {
        acc[j] = __builtin_amdgcn_wmma_f32_16x16x32_bf16(
            false, a, false, bf[j], (short)0, acc[j], false, false);
      }
      a = anext;
      p += 512;
    }
    __syncthreads();  // all waves done reading old x|h

    // ---------------- phase 2: cell update (registers only) ----------
#pragma unroll
    for (int tt = 0; tt < 2; ++tt) {
      int col = (wave * 2 + tt) * 16 + lcol;
#pragma unroll
      for (int r = 0; r < 8; ++r) {
        float iv = sigm_f(acc[0 + tt][r]);
        float fv = sigm_f(acc[2 + tt][r]);
        float gv = tanh_f(acc[4 + tt][r]);
        float ov = sigm_f(acc[6 + tt][r]);
        float c  = fv * creg[tt][r] + iv * gv;
        creg[tt][r] = c;
        float h = ov * tanh_f(c);
        xh[r + rowOff][512 + col] = (bf16_t)h;  // publish h (bf16)
      }
    }
    __syncthreads();  // h visible to all waves

    // ---------------- phase 3: y = h @ W_last^T + b ------------------
    v8f yacc[2];
#pragma unroll
    for (int j = 0; j < 2; ++j)
#pragma unroll
      for (int r = 0; r < 8; ++r) yacc[j][r] = ybias[j];

    const bf16_t* q = lbase;
    v16bf ya = LOAD_A(512);
#pragma unroll 2
    for (int kt = 0; kt < LAST_KT; ++kt) {
      v16bf bf2[2];
#pragma unroll
      for (int j = 0; j < 2; ++j)
        bf2[j] = *(const v16bf*)(q + j * (LAST_KT * 512));
      v16bf yan = LOAD_A(512 + ((kt + 1) & (LAST_KT - 1)) * 32);
#pragma unroll
      for (int j = 0; j < 2; ++j) {
        yacc[j] = __builtin_amdgcn_wmma_f32_16x16x32_bf16(
            false, ya, false, bf2[j], (short)0, yacc[j], false, false);
      }
      ya = yan;
      q += 512;
    }

    // write y to output (fp32) and feed back as next x (bf16)
#pragma unroll
    for (int j = 0; j < 2; ++j) {
      int col = (wave * 2 + j) * 16 + lcol;
#pragma unroll
      for (int r = 0; r < 8; ++r) {
        float y = yacc[j][r];
        out[(size_t)(b0 + r + rowOff) * OLEN * HSZ + (size_t)t * HSZ + col] = y;
        xh[r + rowOff][col] = (bf16_t)y;
      }
    }
    __syncthreads();  // x writes visible before next step's phase 1
  }
#undef LOAD_A
}

// ---------------------------------------------------------------------------
extern "C" void kernel_launch(void* const* d_in, const int* in_sizes, int n_in,
                              void* d_out, int out_size, void* d_ws,
                              size_t ws_size, hipStream_t stream) {
  const float* z     = (const float*)d_in[0];
  const float* Wih   = (const float*)d_in[1];
  const float* Whh   = (const float*)d_in[2];
  const float* bih   = (const float*)d_in[3];
  const float* bhh   = (const float*)d_in[4];
  const float* Wlast = (const float*)d_in[5];
  const float* blast = (const float*)d_in[6];
  float*  out = (float*)d_out;
  bf16_t* ws  = (bf16_t*)d_ws;  // needs 4.5 MB (2,359,296 bf16)

  const int total = GATE_TILE_ELEMS + LAST_TILE_ELEMS;
  lstm_pretile_weights<<<(total + 255) / 256, 256, 0, stream>>>(Wih, Whh,
                                                               Wlast, ws);
  lstm_decode_persistent<<<BSZ / 16, 512, 0, stream>>>(z, bih, bhh, blast, ws,
                                                       out);
}
